// EncoderCell_52329881534792
// MI455X (gfx1250) — matrix operand order repaired and checked by reading, not compiled, not run on hardware
//
#include <hip/hip_runtime.h>
#include <hip/hip_bf16.h>
#include <math.h>

// ---------------------------------------------------------------------------
// Transformer encoder cell for MI455X (gfx1250, wave32, WMMA bf16 path).
// All matmuls use v_wmma_f32_16x16x32_bf16 (bf16 in, f32 accumulate).
// Weights are pre-converted to bf16 and pre-transposed ONCE so the GEMM
// k-loop does pure packed bf16 copies (no per-tile conversion VALU, half the
// global bytes); activations consumed only by GEMMs are kept in bf16.
// ---------------------------------------------------------------------------

typedef __attribute__((ext_vector_type(16))) __bf16 v16bf;
typedef __attribute__((ext_vector_type(4)))  __bf16 v4bf;
typedef __attribute__((ext_vector_type(8)))  float  v8f;

#define D_MODEL 1024
#define NHEAD   16
#define DHEAD   64
#define SEQ     2048
#define BATCH   2
#define FFH     4096
#define MROWS   (BATCH * SEQ)   // 4096 token rows
#define LN_EPS  1e-5f

__device__ __forceinline__ v8f wmma_bf16(v16bf a, v16bf b, v8f c) {
  // (neg_a, A, neg_b, B, c_mod, C, reuse_a, reuse_b)
  return __builtin_amdgcn_wmma_f32_16x16x32_bf16(false, a, false, b,
                                                 (short)0, c, false, false);
}

// A-fragment (16 rows x 32 K, bf16) from row-major LDS tile [rows][ld].
// Layout: lanes 0-15 hold K 0..15, lanes 16-31 hold K 16..31 (matches B frags);
// each 8-element run is contiguous -> two 16B LDS reads per lane.
__device__ __forceinline__ v16bf load_a_frag(const __bf16* tile, int ld,
                                             int row0, int k0, int lane) {
  int l16 = lane & 15, h = (lane >> 4) & 1;
  const __bf16* p0 = tile + (row0 + l16) * ld + k0 + h * 16;
  const __bf16* p1 = p0 + 8;
  v16bf a;
#pragma unroll
  for (int i = 0; i < 8; ++i) { a[i] = p0[i]; a[i + 8] = p1[i]; }
  return a;
}

// B-fragment (32 K x 16 cols, bf16) from LDS tile stored TRANSPOSED: [col][ld]
// with K contiguous. Contiguous k (as in A frags): lanes 0-15 = K 0..15, lanes 16-31 = K 16..31,
// column = lane%16 -> one 32B contiguous read per lane.
__device__ __forceinline__ v16bf load_bt_frag(const __bf16* tile, int ld,
                                              int col0, int k0, int lane) {
  int l16 = lane & 15, h = (lane >> 4) & 1;
  const __bf16* p = tile + (col0 + l16) * ld + k0 + h * 16;
  v16bf b;
#pragma unroll
  for (int i = 0; i < 16; ++i) b[i] = p[i];
  return b;
}

// ---------------------------------------------------------------------------
// One-time weight prep: W[K,N] f32 -> WT[N,K] bf16 (optionally scaled).
// 32x32 LDS tile so both the f32 reads and bf16 writes are coalesced.
// ---------------------------------------------------------------------------
__global__ __launch_bounds__(256) void transpose_convert_kernel(
    const float* __restrict__ W, __bf16* __restrict__ WT,
    int K, int N, float scale) {
  __shared__ float t[32][33];
  const int k0 = blockIdx.y * 32, n0 = blockIdx.x * 32;
  const int tx = threadIdx.x & 31, ty = threadIdx.x >> 5;  // 32 x 8
#pragma unroll
  for (int p = 0; p < 4; ++p) {
    int kk = ty + p * 8;
    t[kk][tx] = W[(size_t)(k0 + kk) * N + n0 + tx];
  }
  __syncthreads();
#pragma unroll
  for (int p = 0; p < 4; ++p) {
    int nn = ty + p * 8;
    WT[(size_t)(n0 + nn) * K + k0 + tx] = (__bf16)(t[tx][nn] * scale);
  }
}

// Elementwise f32 -> bf16 (packed 8B stores).
__global__ __launch_bounds__(256) void f32_to_bf16_kernel(
    const float* __restrict__ src, __bf16* __restrict__ dst) {
  size_t i = ((size_t)blockIdx.x * 256 + threadIdx.x) * 4;
  float4 f = *(const float4*)(src + i);
  v4bf o;
  o[0] = (__bf16)f.x; o[1] = (__bf16)f.y; o[2] = (__bf16)f.z; o[3] = (__bf16)f.w;
  *(v4bf*)(dst + i) = o;
}

// ---------------------------------------------------------------------------
// bf16 WMMA GEMM:  C[M,N] = act(A[M,K] @ B[K,N] (+ bias))
// A: bf16 row-major [M,K].  BT: bf16 pre-transposed [N,K].
// Block tile 128x128, k-step 32, 8 waves: 2(M) x 4(N), each wave 64x32.
// Staging is pure packed bf16 copies (ds_store_b64); next k-tile prefetched.
// ---------------------------------------------------------------------------
template <bool HAS_BIAS, bool RELU, bool OUT_BF16>
__global__ __launch_bounds__(256) void gemm_bf16_kernel(
    const __bf16* __restrict__ A, const __bf16* __restrict__ BT,
    const float* __restrict__ bias, void* __restrict__ Cout,
    int M, int N, int K) {
  constexpr int BM = 128, BN = 128, BK = 32, LDT = BK + 8;
  __shared__ __bf16 lA[BM * LDT];   // [m][k]
  __shared__ __bf16 lB[BN * LDT];   // [n][k]

  const int tid = threadIdx.x, lane = tid & 31, w = tid >> 5;
  const int wm = (w >> 2) * 64;     // wave M offset: 0 / 64
  const int wn = (w & 3) * 32;      // wave N offset: 0 / 32 / 64 / 96
  const int m0 = blockIdx.y * BM, n0 = blockIdx.x * BN;

  const v8f zero8 = {0.f, 0.f, 0.f, 0.f, 0.f, 0.f, 0.f, 0.f};
  v8f acc[4][2];
#pragma unroll
  for (int i = 0; i < 4; ++i)
#pragma unroll
    for (int j = 0; j < 2; ++j) acc[i][j] = zero8;

  for (int k0 = 0; k0 < K; k0 += BK) {
    // Stage A and BT tiles: 128x32 bf16 each = 1024 v4bf units apiece.
#pragma unroll
    for (int p = 0; p < 4; ++p) {
      int idx = p * 256 + tid;
      int r = idx >> 3, c = (idx & 7) << 2;
      *(v4bf*)(lA + r * LDT + c) =
          *(const v4bf*)(A + (size_t)(m0 + r) * K + k0 + c);
      *(v4bf*)(lB + r * LDT + c) =
          *(const v4bf*)(BT + (size_t)(n0 + r) * K + k0 + c);
    }
    // Prefetch next k-tile into cache while WMMAs run (global_prefetch_b8).
    if (k0 + BK < K) {
      int r = tid >> 3, c = (tid & 7) << 2;
      __builtin_prefetch(A + (size_t)(m0 + r) * K + k0 + BK + c, 0, 3);
      __builtin_prefetch(BT + (size_t)(n0 + r) * K + k0 + BK + c, 0, 3);
    }
    __syncthreads();

    v16bf af[4], bf2[2];
#pragma unroll
    for (int i = 0; i < 4; ++i) af[i] = load_a_frag(lA, LDT, wm + 16 * i, 0, lane);
#pragma unroll
    for (int j = 0; j < 2; ++j) bf2[j] = load_bt_frag(lB, LDT, wn + 16 * j, 0, lane);
#pragma unroll
    for (int i = 0; i < 4; ++i)
#pragma unroll
      for (int j = 0; j < 2; ++j)
        acc[i][j] = wmma_bf16(af[i], bf2[j], acc[i][j]);
    __syncthreads();
  }

  // Epilogue. C layout: VGPR r -> row (m0+wm+16i + 8h + r), col = n0+wn+16j+lane%16
  const int l16 = lane & 15, h = lane >> 4;
  float* Cf = (float*)Cout;
  __bf16* Cb = (__bf16*)Cout;
#pragma unroll
  for (int i = 0; i < 4; ++i)
#pragma unroll
    for (int j = 0; j < 2; ++j) {
      int col = n0 + wn + 16 * j + l16;
      float bv = HAS_BIAS ? bias[col] : 0.f;
#pragma unroll
      for (int r = 0; r < 8; ++r) {
        int row = m0 + wm + 16 * i + (h ? 8 : 0) + r;
        float val = acc[i][j][r] + bv;
        if (RELU) val = val > 0.f ? val : 0.f;
        if (OUT_BF16) Cb[(size_t)row * N + col] = (__bf16)val;
        else          Cf[(size_t)row * N + col] = val;
      }
    }
}

// ---------------------------------------------------------------------------
// Flash-style attention (no score materialization), bf16 in / bf16 out.
// Computes S^T = K.Q^T so the per-query softmax reduction is per-lane + one
// shfl_xor(16); P^T is repacked to B-fragments in-registers (one half-wave
// swap) and Z^T = V^T.P^T accumulates with online rescale. 64-key chunks.
// 1/sqrt(DH) is pre-folded into WqT, so Qp arrives pre-scaled.
// Grid: (SEQ/128, BATCH*NHEAD); block 256 = 8 waves, 16 query rows per wave.
// ---------------------------------------------------------------------------
__global__ __launch_bounds__(256) void attention_kernel(
    const __bf16* __restrict__ Qp, const __bf16* __restrict__ Kp,
    const __bf16* __restrict__ Vp, const float* __restrict__ mask,
    __bf16* __restrict__ Z) {
  constexpr int KC  = 64;          // keys per chunk
  constexpr int LDQ = DHEAD + 8;   // 72
  constexpr int LDK = DHEAD + 8;
  constexpr int LDV = KC + 8;      // keys-contiguous, padded
  __shared__ __bf16 lQ[128 * LDQ];      // [q][dh] (pre-scaled)
  __shared__ __bf16 lK[KC * LDK];       // [key][dh]
  __shared__ __bf16 lVT[DHEAD * LDV];   // [dh][key] (transposed V chunk)

  const int tid = threadIdx.x, lane = tid & 31, w = tid >> 5;
  const int l16 = lane & 15, h = lane >> 4;
  const int bh = blockIdx.y;
  const int b = bh / NHEAD, hh = bh % NHEAD;
  const int qBlock = blockIdx.x * 128;
  const size_t rowBase = (size_t)b * SEQ;
  const int colBase = hh * DHEAD;

  // Stage Q tile once (pure bf16 copy).
#pragma unroll
  for (int p = 0; p < 8; ++p) {
    int idx = p * 256 + tid;                 // 2048 v4bf units
    int r = idx >> 4, c = (idx & 15) << 2;
    *(v4bf*)(lQ + r * LDQ + c) =
        *(const v4bf*)(Qp + (rowBase + qBlock + r) * D_MODEL + colBase + c);
  }
  __syncthreads();

  // Per-wave Q^T B-fragments (k-dim = dh, two 32-chunks), kept in registers.
  const v16bf qb0 = load_bt_frag(lQ, LDQ, w * 16, 0, lane);
  const v16bf qb1 = load_bt_frag(lQ, LDQ, w * 16, 32, lane);
  const int q0 = qBlock + w * 16;

  const v8f zero8 = {0.f, 0.f, 0.f, 0.f, 0.f, 0.f, 0.f, 0.f};
  v8f acc[4];                 // Z^T: 4 x (16 dh x 16 q)
#pragma unroll
  for (int i = 0; i < 4; ++i) acc[i] = zero8;
  float mrun = -3.0e38f, lrun = 0.f;

  for (int ks = 0; ks < SEQ; ks += KC) {
    __syncthreads();          // previous chunk's LDS reads complete
    // Stage K chunk [64 keys][64 dh] and V chunk transposed [64 dh][64 keys].
#pragma unroll
    for (int p = 0; p < 4; ++p) {
      int idx = p * 256 + tid;               // 1024 v4bf units
      int r = idx >> 4, c = (idx & 15) << 2;
      *(v4bf*)(lK + r * LDK + c) =
          *(const v4bf*)(Kp + (rowBase + ks + r) * D_MODEL + colBase + c);
      v4bf g = *(const v4bf*)(Vp + (rowBase + ks + r) * D_MODEL + colBase + c);
      lVT[(c + 0) * LDV + r] = g[0];
      lVT[(c + 1) * LDV + r] = g[1];
      lVT[(c + 2) * LDV + r] = g[2];
      lVT[(c + 3) * LDV + r] = g[3];
    }
    __syncthreads();

    // S^T fragments: pf[f] = keys 16f..16f+15 (M=key, N=query).
    v8f pf[4];
#pragma unroll
    for (int f = 0; f < 4; ++f) {
      pf[f] = zero8;
      v16bf ka;
      ka = load_a_frag(lK, LDK, f * 16,  0, lane); pf[f] = wmma_bf16(ka, qb0, pf[f]);
      ka = load_a_frag(lK, LDK, f * 16, 32, lane); pf[f] = wmma_bf16(ka, qb1, pf[f]);
    }

    // Additive mask (broadcast over b,h): mask[q][key]
    const float* mrow = mask + (size_t)(q0 + l16) * SEQ + ks + (h ? 8 : 0);
#pragma unroll
    for (int f = 0; f < 4; ++f)
#pragma unroll
      for (int r = 0; r < 8; ++r) pf[f][r] += mrow[16 * f + r];

    // Online softmax along keys: per lane = fixed query column.
    float cmax = -3.0e38f;
#pragma unroll
    for (int f = 0; f < 4; ++f)
#pragma unroll
      for (int r = 0; r < 8; ++r) cmax = fmaxf(cmax, pf[f][r]);
    cmax = fmaxf(cmax, __shfl_xor(cmax, 16, 32));
    float mnew = fmaxf(mrun, cmax);
    float corr = __expf(mrun - mnew);
    mrun = mnew;
    float ls = 0.f;
#pragma unroll
    for (int f = 0; f < 4; ++f)
#pragma unroll
      for (int r = 0; r < 8; ++r) {
        pf[f][r] = __expf(pf[f][r] - mnew);
        ls += pf[f][r];
      }
    ls += __shfl_xor(ls, 16, 32);
    lrun = lrun * corr + ls;
#pragma unroll
    for (int i = 0; i < 4; ++i)
#pragma unroll
      for (int e = 0; e < 8; ++e) acc[i][e] *= corr;

    // Repack P^T (C-layout) into two 32x16 B-fragments via half-wave swaps.
    v16bf pb[2];
#pragma unroll
    for (int g2 = 0; g2 < 2; ++g2) {
      const v8f& lo = pf[2 * g2];      // keys base+0..15
      const v8f& hi = pf[2 * g2 + 1];  // keys base+16..31
#pragma unroll
      for (int r = 0; r < 8; ++r) {
        float x0 = __shfl_xor(lo[r], 16, 32);
        float x1 = __shfl_xor(hi[r], 16, 32);
        pb[g2][r]     = (__bf16)(h ? x1    : lo[r]);   // keys h*16 + r
        pb[g2][r + 8] = (__bf16)(h ? hi[r] : x0);      // keys h*16 + 8 + r
      }
    }

    // Z^T += V^T (16 dh x 64 keys) @ P^T (64 keys x 16 q)
#pragma unroll
    for (int i = 0; i < 4; ++i) {
      v16bf va;
      va = load_a_frag(lVT, LDV, 16 * i,  0, lane); acc[i] = wmma_bf16(va, pb[0], acc[i]);
      va = load_a_frag(lVT, LDV, 16 * i, 32, lane); acc[i] = wmma_bf16(va, pb[1], acc[i]);
    }
  }

  // Normalize and scatter back as bf16 (8 contiguous elems = two 8B stores).
  const float linv = 1.f / lrun;
  const int qrow = q0 + l16;
#pragma unroll
  for (int i = 0; i < 4; ++i) {
    __bf16* dst = Z + (rowBase + qrow) * D_MODEL + colBase + 16 * i + (h ? 8 : 0);
    v4bf o0, o1;
#pragma unroll
    for (int r = 0; r < 4; ++r) {
      o0[r] = (__bf16)(acc[i][r] * linv);
      o1[r] = (__bf16)(acc[i][r + 4] * linv);
    }
    *(v4bf*)(dst) = o0;
    *(v4bf*)(dst + 4) = o1;
  }
}

// ---------------------------------------------------------------------------
// LayerNorm over D_MODEL.
//   ADD_PRE=true : out = LN(src + src2)*g + b          (final LN)
//   ADD_PRE=false: out = LN(src)*g + b + src2          (post-attn LN + resid q)
// EMIT_BF16: also write a bf16 copy (feeds the next GEMM's A side).
// ---------------------------------------------------------------------------
template <bool ADD_PRE, bool EMIT_BF16>
__global__ __launch_bounds__(256) void ln_kernel(
    const float* __restrict__ src, const float* __restrict__ src2,
    const float* __restrict__ gamma, const float* __restrict__ beta,
    float* __restrict__ out, __bf16* __restrict__ out_bf) {
  const int row = blockIdx.x, tid = threadIdx.x;
  const float* x  = src  + (size_t)row * D_MODEL;
  const float* x2 = src2 + (size_t)row * D_MODEL;
  __shared__ float red1[256], red2[256];

  float vals[4];
  float s = 0.f, sq = 0.f;
#pragma unroll
  for (int i = 0; i < 4; ++i) {
    int c = i * 256 + tid;
    float t = x[c];
    if (ADD_PRE) t += x2[c];
    vals[i] = t;
    s += t; sq += t * t;
  }
  red1[tid] = s; red2[tid] = sq;
  __syncthreads();
  for (int off = 128; off > 0; off >>= 1) {
    if (tid < off) { red1[tid] += red1[tid + off]; red2[tid] += red2[tid + off]; }
    __syncthreads();
  }
  const float mu  = red1[0] * (1.f / D_MODEL);
  const float var = red2[0] * (1.f / D_MODEL) - mu * mu;
  const float rstd = rsqrtf(var + LN_EPS);
#pragma unroll
  for (int i = 0; i < 4; ++i) {
    int c = i * 256 + tid;
    float y = (vals[i] - mu) * rstd * gamma[c] + beta[c];
    if (!ADD_PRE) y += x2[c];
    out[(size_t)row * D_MODEL + c] = y;
    if (EMIT_BF16) out_bf[(size_t)row * D_MODEL + c] = (__bf16)y;
  }
}

// ---------------------------------------------------------------------------
extern "C" void kernel_launch(void* const* d_in, const int* in_sizes, int n_in,
                              void* d_out, int out_size, void* d_ws, size_t ws_size,
                              hipStream_t stream) {
  (void)in_sizes; (void)n_in; (void)out_size; (void)ws_size;
  const float* q    = (const float*)d_in[0];
  const float* k    = (const float*)d_in[1];
  const float* v    = (const float*)d_in[2];
  /* d_in[3] = time_lengths: present in the signature but unused by the math */
  const float* mask = (const float*)d_in[4];
  const float* Wq   = (const float*)d_in[5];
  const float* Wk   = (const float*)d_in[6];
  const float* Wv   = (const float*)d_in[7];
  const float* Wd   = (const float*)d_in[8];
  const float* W1   = (const float*)d_in[9];
  const float* b1   = (const float*)d_in[10];
  const float* W2   = (const float*)d_in[11];
  const float* b2   = (const float*)d_in[12];
  const float* g1   = (const float*)d_in[13];
  const float* be1  = (const float*)d_in[14];
  const float* g2   = (const float*)d_in[15];
  const float* be2  = (const float*)d_in[16];
  float* out = (float*)d_out;
  float* ws  = (float*)d_ws;

  // Workspace layout in units of 1M floats (4 MB); bf16 buffers use elems/2.
  const size_t U = 1ull << 20;
  __bf16* qb  = (__bf16*)(ws +  0 * U);   // 4M elems
  __bf16* kb  = (__bf16*)(ws +  2 * U);
  __bf16* vb  = (__bf16*)(ws +  4 * U);
  __bf16* WqT = (__bf16*)(ws +  6 * U);   // 1M elems, pre-scaled by 1/8
  __bf16* WkT = (__bf16*)(ws +  6 * U + U / 2);
  __bf16* WvT = (__bf16*)(ws +  7 * U);
  __bf16* WdT = (__bf16*)(ws +  7 * U + U / 2);
  __bf16* W1T = (__bf16*)(ws +  8 * U);   // 4M elems
  __bf16* W2T = (__bf16*)(ws + 10 * U);   // 4M elems
  __bf16* Qp  = (__bf16*)(ws + 12 * U);   // 4M elems
  __bf16* Kp  = (__bf16*)(ws + 14 * U);
  __bf16* Vp  = (__bf16*)(ws + 16 * U);
  __bf16* Zb  = (__bf16*)(ws + 18 * U);   // attention out (pre-Wd)
  float*  Zd  =           ws + 20 * U;    // z @ Wd (f32, feeds LN1)
  float*  Xb  =           ws + 24 * U;    // LN1(zd)+q (f32, LN2 residual)
  __bf16* Xbb = (__bf16*)(ws + 28 * U);   // bf16 copy feeding FF1
  __bf16* Hb  = (__bf16*)(ws + 30 * U);   // relu(x@W1+b1): 16M elems
  float*  F2  =           ws + 38 * U;    // h@W2+b2 (f32, feeds LN2)

  const dim3 blk(256);
  const dim3 gP(D_MODEL / 128, MROWS / 128);   // (8, 32)
  const dim3 gF1(FFH / 128,    MROWS / 128);   // (32, 32)
  const dim3 gA(SEQ / 128, BATCH * NHEAD);     // (16, 32)
  const dim3 gLN(MROWS);
  const dim3 gCvt((MROWS * D_MODEL) / 1024);   // 4096 blocks
  const dim3 gTd(D_MODEL / 32, D_MODEL / 32);  // DxD transpose
  const dim3 gT1(FFH / 32,     D_MODEL / 32);  // W1 [1024,4096]
  const dim3 gT2(D_MODEL / 32, FFH / 32);      // W2 [4096,1024]

  // One-time precision prep (activations + weights; 1/sqrt(DH) folded into Wq).
  f32_to_bf16_kernel<<<gCvt, blk, 0, stream>>>(q, qb);
  f32_to_bf16_kernel<<<gCvt, blk, 0, stream>>>(k, kb);
  f32_to_bf16_kernel<<<gCvt, blk, 0, stream>>>(v, vb);
  transpose_convert_kernel<<<gTd, blk, 0, stream>>>(Wq, WqT, D_MODEL, D_MODEL, 0.125f);
  transpose_convert_kernel<<<gTd, blk, 0, stream>>>(Wk, WkT, D_MODEL, D_MODEL, 1.f);
  transpose_convert_kernel<<<gTd, blk, 0, stream>>>(Wv, WvT, D_MODEL, D_MODEL, 1.f);
  transpose_convert_kernel<<<gTd, blk, 0, stream>>>(Wd, WdT, D_MODEL, D_MODEL, 1.f);
  transpose_convert_kernel<<<gT1, blk, 0, stream>>>(W1, W1T, D_MODEL, FFH, 1.f);
  transpose_convert_kernel<<<gT2, blk, 0, stream>>>(W2, W2T, FFH, D_MODEL, 1.f);

  // Q/K/V projections (bf16 out; Qp arrives pre-scaled by 1/8).
  gemm_bf16_kernel<false, false, true><<<gP, blk, 0, stream>>>(qb, WqT, nullptr, Qp, MROWS, D_MODEL, D_MODEL);
  gemm_bf16_kernel<false, false, true><<<gP, blk, 0, stream>>>(kb, WkT, nullptr, Kp, MROWS, D_MODEL, D_MODEL);
  gemm_bf16_kernel<false, false, true><<<gP, blk, 0, stream>>>(vb, WvT, nullptr, Vp, MROWS, D_MODEL, D_MODEL);
  // Multi-head attention (flash-style, WMMA).
  attention_kernel<<<gA, blk, 0, stream>>>(Qp, Kp, Vp, mask, Zb);
  // Output projection (f32 out, feeds LN1).
  gemm_bf16_kernel<false, false, false><<<gP, blk, 0, stream>>>(Zb, WdT, nullptr, Zd, MROWS, D_MODEL, D_MODEL);
  // x = LN1(zd)*g1+be1 + q   (f32 + bf16 copy for FF1)
  ln_kernel<false, true><<<gLN, blk, 0, stream>>>(Zd, q, g1, be1, Xb, Xbb);
  // FF
  gemm_bf16_kernel<true, true,  true ><<<gF1, blk, 0, stream>>>(Xbb, W1T, b1, Hb, MROWS, FFH, D_MODEL);
  gemm_bf16_kernel<true, false, false><<<gP,  blk, 0, stream>>>(Hb,  W2T, b2, F2, MROWS, D_MODEL, FFH);
  // out = LN2(ff + x)*g2+be2
  ln_kernel<true, false><<<gLN, blk, 0, stream>>>(F2, Xb, g2, be2, out, nullptr);
}
